// GATHeadLayer_17171279249900
// MI455X (gfx1250) — matrix operand order repaired and checked by reading, not compiled, not run on hardware
//
#include <hip/hip_runtime.h>
#include <math.h>

#define N_NODES 100000
#define E_EDGES 1600000
#define IN_DIM  128
#define OUT_DIM 64
#define ALPHA   0.2f

typedef __attribute__((ext_vector_type(16))) _Float16 v16h;
typedef __attribute__((ext_vector_type(8)))  float    v8f;
typedef __attribute__((ext_vector_type(4)))  float    v4f;

// ---------------------------------------------------------------------------
// Kernel 0: one-time W_fc f32 -> f16 conversion (16 KB, stays cache-hot).
// Removes 128 v_cvt_pk_f16_f32 per GEMM wave and halves B-fragment bytes.
// ---------------------------------------------------------------------------
__global__ void gat_wcvt_kernel(const float* __restrict__ Wfc,
                                _Float16* __restrict__ W16) {
  const int i = blockIdx.x * blockDim.x + threadIdx.x;
  if (i < OUT_DIM * IN_DIM) W16[i] = (_Float16)Wfc[i];
}

// ---------------------------------------------------------------------------
// Kernel 1: h = x @ W_fc^T   via v_wmma_f32_16x16x32_f16
// One wave -> one 16-row tile of h (16 x 64), 4 n-tiles x 4 k-steps = 16 WMMAs.
// ---------------------------------------------------------------------------
__global__ __launch_bounds__(256) void gat_gemm_kernel(
    const float* __restrict__ x,
    const _Float16* __restrict__ W16,
    float* __restrict__ h) {
  const int lane  = threadIdx.x & 31;
  const int wave  = threadIdx.x >> 5;
  const int tile  = blockIdx.x * 8 + wave;
  if (tile * 16 >= N_NODES) return;

  const int row0  = tile * 16;
  const int mlane = lane & 15;
  const int hi    = lane >> 4;          // 0: lanes 0-15, 1: lanes 16-31

  v8f acc[4] = {v8f{}, v8f{}, v8f{}, v8f{}};

  const int arow = row0 + mlane;
  const int aoff = hi * 8;              // A: hi lanes take K sub-block +8
  const int boff = hi * 8;              // B: hi lanes take K sub-block +8 (same as A)

  const float* xrow = x + (size_t)arow * IN_DIM;

  #pragma unroll
  for (int k0 = 0; k0 < IN_DIM; k0 += 32) {
    // Prefetch next K chunk of this row (gfx1250 global_prefetch_b8)
    if (k0 + 32 < IN_DIM)
      __builtin_prefetch(xrow + k0 + 32 + aoff, 0, 1);

    // ---- A fragment: 16x32 f16, ISA layout (lane<16: K 0..7,16..23) ----
    const float* px = xrow + k0 + aoff;
    v4f a0 = *(const v4f*)(px + 0);
    v4f a1 = *(const v4f*)(px + 4);
    v4f a2 = *(const v4f*)(px + 16);
    v4f a3 = *(const v4f*)(px + 20);
    v16h afrag;
    #pragma unroll
    for (int j = 0; j < 4; ++j) {
      afrag[j]      = (_Float16)a0[j];
      afrag[4 + j]  = (_Float16)a1[j];
      afrag[8 + j]  = (_Float16)a2[j];
      afrag[12 + j] = (_Float16)a3[j];
    }

    // ---- 4 N-tiles: B = W^T, B[k][n] = W16[n][k]; 8 halves at +0 and 8 at +16 (same k order as A) ----
    #pragma unroll
    for (int t = 0; t < 4; ++t) {
      const int col = t * 16 + mlane;   // output channel handled by this lane
      typedef __attribute__((ext_vector_type(8))) _Float16 v8h_t;
      const _Float16* pw = W16 + (size_t)col * IN_DIM + k0 + boff;
      const v8h_t b0 = *(const v8h_t*)(pw + 0);
      const v8h_t b1 = *(const v8h_t*)(pw + 16);
      const v16h bfrag = __builtin_shufflevector(b0, b1, 0, 1, 2, 3, 4, 5, 6, 7, 8, 9, 10, 11, 12, 13, 14, 15);
      // D = A*B + C  (f32 accum)
      acc[t] = __builtin_amdgcn_wmma_f32_16x16x32_f16(
          false, afrag, false, bfrag, (short)0, acc[t], false, false);
    }
  }

  // ---- store C/D: VGPR r -> row (r + 8*hi), col (16*t + mlane) ----
  #pragma unroll
  for (int t = 0; t < 4; ++t) {
    #pragma unroll
    for (int r = 0; r < 8; ++r) {
      h[(size_t)(row0 + r + 8 * hi) * OUT_DIM + t * 16 + mlane] = acc[t][r];
    }
  }
}

// ---------------------------------------------------------------------------
// Kernel 2: per-node attention scores  s1 = h@a1, s2 = h@a2  (one wave/row)
// ---------------------------------------------------------------------------
__global__ __launch_bounds__(256) void gat_score_kernel(
    const float* __restrict__ h,
    const float* __restrict__ Wattn,
    float* __restrict__ score1,
    float* __restrict__ score2) {
  const int lane = threadIdx.x & 31;
  const int row  = blockIdx.x * 8 + (threadIdx.x >> 5);
  if (row >= N_NODES) return;
  const float h0 = h[(size_t)row * OUT_DIM + lane];
  const float h1 = h[(size_t)row * OUT_DIM + lane + 32];
  float s1 = h0 * Wattn[lane]      + h1 * Wattn[lane + 32];
  float s2 = h0 * Wattn[64 + lane] + h1 * Wattn[96 + lane];
  #pragma unroll
  for (int m = 16; m >= 1; m >>= 1) {
    s1 += __shfl_xor(s1, m, 32);
    s2 += __shfl_xor(s2, m, 32);
  }
  if (lane == 0) { score1[row] = s1; score2[row] = s2; }
}

// ---------------------------------------------------------------------------
// Kernel 3: zero accumulators (ws is poisoned by the harness)
// ---------------------------------------------------------------------------
__global__ void gat_zero_kernel(float* __restrict__ hprime,
                                float* __restrict__ rowsum) {
  const int i = blockIdx.x * blockDim.x + threadIdx.x;
  if (i < N_NODES * OUT_DIM) hprime[i] = 0.0f;
  if (i < N_NODES)           rowsum[i] = 0.0f;
}

// ---------------------------------------------------------------------------
// Kernel 4: edge scatter. One wave per edge: 32 lanes x 2 channels.
// edge_e = exp(-leaky_relu(s1[src]+s2[dst])); hprime[src] += edge_e * h[dst]
// h / hprime are L2-resident (25.6 MB each, 192 MB L2).
// ---------------------------------------------------------------------------
__global__ __launch_bounds__(256) void gat_edge_kernel(
    const int* __restrict__ ei,
    const float* __restrict__ h,
    const float* __restrict__ score1,
    const float* __restrict__ score2,
    float* __restrict__ hprime,
    float* __restrict__ rowsum) {
  const int lane = threadIdx.x & 31;
  const int e    = blockIdx.x * 8 + (threadIdx.x >> 5);
  if (e >= E_EDGES) return;
  const int s = ei[e];
  const int d = ei[E_EDGES + e];
  const float sc = score1[s] + score2[d];
  const float lr = sc > 0.0f ? sc : ALPHA * sc;
  const float w  = __expf(-lr);
  const float v0 = h[(size_t)d * OUT_DIM + lane];
  const float v1 = h[(size_t)d * OUT_DIM + lane + 32];
  atomicAdd(&hprime[(size_t)s * OUT_DIM + lane],      w * v0);
  atomicAdd(&hprime[(size_t)s * OUT_DIM + lane + 32], w * v1);
  if (lane == 0) atomicAdd(&rowsum[s], w);
}

// ---------------------------------------------------------------------------
// Kernel 5: out = elu(hprime / rowsum * n_norm)
// ---------------------------------------------------------------------------
__global__ void gat_finalize_kernel(const float* __restrict__ hprime,
                                    const float* __restrict__ rowsum,
                                    const float* __restrict__ nnorm,
                                    float* __restrict__ out) {
  const int i = blockIdx.x * blockDim.x + threadIdx.x;
  if (i >= N_NODES * OUT_DIM) return;
  const int row = i >> 6;
  const float v = hprime[i] / rowsum[row] * nnorm[row];
  out[i] = v > 0.0f ? v : (__expf(v) - 1.0f);
}

extern "C" void kernel_launch(void* const* d_in, const int* in_sizes, int n_in,
                              void* d_out, int out_size, void* d_ws, size_t ws_size,
                              hipStream_t stream) {
  const float* x     = (const float*)d_in[0];  // N x 128
  const int*   ei    = (const int*)  d_in[1];  // 2 x E
  const float* nnorm = (const float*)d_in[2];  // N x 1
  const float* Wfc   = (const float*)d_in[3];  // 64 x 128
  const float* Wattn = (const float*)d_in[4];  // 1 x 128
  float* out = (float*)d_out;                  // N x 64

  // Workspace layout (f32): h | hprime | score1 | score2 | rowsum | W16(f16)
  float* h   = (float*)d_ws;
  float* hp  = h  + (size_t)N_NODES * OUT_DIM;
  float* s1  = hp + (size_t)N_NODES * OUT_DIM;
  float* s2  = s1 + N_NODES;
  float* rs  = s2 + N_NODES;
  _Float16* W16 = (_Float16*)(rs + N_NODES);   // 64*128 halves (32B-aligned)

  gat_wcvt_kernel<<<(OUT_DIM * IN_DIM + 255) / 256, 256, 0, stream>>>(Wfc, W16);
  gat_zero_kernel<<<(N_NODES * OUT_DIM + 255) / 256, 256, 0, stream>>>(hp, rs);
  gat_gemm_kernel<<<((N_NODES / 16) + 7) / 8, 256, 0, stream>>>(x, W16, h);
  gat_score_kernel<<<(N_NODES + 7) / 8, 256, 0, stream>>>(h, Wattn, s1, s2);
  gat_edge_kernel<<<(E_EDGES + 7) / 8, 256, 0, stream>>>(ei, h, s1, s2, hp, rs);
  gat_finalize_kernel<<<(N_NODES * OUT_DIM + 255) / 256, 256, 0, stream>>>(hp, rs, nnorm, out);
}
